// SimpleLSTM_58789512347929
// MI455X (gfx1250) — matrix operand order repaired and checked by reading, not compiled, not run on hardware
//
#include <hip/hip_runtime.h>
#include <hip/hip_fp16.h>

// ---------------- problem constants ----------------
#define SEQ    2048
#define EDIM   1024
#define VOCAB  50257
#define GATE4  4096                  // 4*EDIM
#define KTILES 32                    // EDIM / 32
#define MT_SEQ (SEQ/16)              // 128 m-tiles
#define NT_WIH (GATE4/16)            // 256 n-tiles
#define NT_HEAD ((VOCAB+15)/16)      // 3142 n-tiles (padded)
#define TILE_HALVES 512              // one 16x32 (A) or 32x16 (B) tile: 32 lanes * 16 halves
#define ABLK (KTILES*TILE_HALVES)    // all-K slab for one 16-row tile: 16384 halves = 32 KB
#define MB 4                         // m-tiles per block (register blocking)

typedef __attribute__((ext_vector_type(16))) _Float16 v16h;
typedef __attribute__((ext_vector_type(8)))  float    v8f;

// (mtile,ktile,m,k) -> flat index in packed A buffer, matching the CDNA5
// 16-bit A-matrix 16x32 VGPR layout (CDNA5 ISA 7.12.2):
//   K = (v<4?0:16) + 2*(v%4) + j + (lane>=16 ? 8 : 0),  M = lane%16
__device__ __forceinline__ int a_pack_index(int mtile, int ktile, int m, int k) {
    int base16 = (k >= 16) ? 1 : 0;
    int k16 = k & 15;
    int hi8 = (k16 >= 8) ? 1 : 0;
    int k8  = k16 & 7;
    int v   = base16 * 4 + (k8 >> 1);
    int j   = k8 & 1;
    int half = 2 * v + j;
    int lane = m + hi8 * 16;
    return (mtile * KTILES + ktile) * TILE_HALVES + lane * 16 + half;
}

// ---------------- packing kernels ----------------

__global__ void k_bias_sum(const float* __restrict__ bih, const float* __restrict__ bhh,
                           float* __restrict__ bsum) {
    int i = blockIdx.x * 256 + threadIdx.x;
    if (i < GATE4) bsum[i] = bih[i] + bhh[i];
}

__global__ void k_pack_embed(const int* __restrict__ sent, const float* __restrict__ wte,
                             _Float16* __restrict__ Ap) {
    int id = blockIdx.x * 256 + threadIdx.x;       // SEQ*EDIM elements
    int t  = id >> 10;
    int kf = id & 1023;
    int tok = sent[t];
    float v = wte[(size_t)tok * EDIM + kf];
    Ap[a_pack_index(t >> 4, kf >> 5, t & 15, kf & 31)] = (_Float16)v;
}

// W [Nreal rows][1024] f32 -> packed B f16, B-matrix 32x16 layout:
//   n = lane%16, k_local = (half/8)*16 + (lane/16)*8 + half%8
__global__ void k_pack_B(const float* __restrict__ W, _Float16* __restrict__ Bp,
                         int Nreal) {
    int id = blockIdx.x * 256 + threadIdx.x;       // ntiles*ABLK elements
    int ntile = id / ABLK;
    int rem   = id % ABLK;
    int kt    = rem / TILE_HALVES;
    int pos   = rem % TILE_HALVES;
    int lane  = pos >> 4;
    int h     = pos & 15;
    int n     = lane & 15;
    int kloc  = ((h >> 3) << 4) + ((lane >> 4) << 3) + (h & 7);
    int row   = ntile * 16 + n;
    int k     = kt * 32 + kloc;
    float v = (row < Nreal) ? W[(size_t)row * EDIM + k] : 0.0f;
    Bp[id] = (_Float16)v;
}

// W_hh [4096][1024] -> Wp[k][r] = float4{ i,f,g,o rows r at column k }
__global__ void k_pack_whh(const float* __restrict__ Whh, float* __restrict__ Wp) {
    int id = blockIdx.x * 256 + threadIdx.x;       // EDIM*EDIM
    int k = id >> 10;
    int r = id & 1023;
    float4 w;
    w.x = Whh[((size_t)(0 * EDIM + r)) * EDIM + k];
    w.y = Whh[((size_t)(1 * EDIM + r)) * EDIM + k];
    w.z = Whh[((size_t)(2 * EDIM + r)) * EDIM + k];
    w.w = Whh[((size_t)(3 * EDIM + r)) * EDIM + k];
    ((float4*)Wp)[(size_t)k * EDIM + r] = w;
}

// ---------------- WMMA GEMM ----------------
// out[16*mtiles, Nreal] = A(f16,packed) x B(f16,packed) + bias, f32 accum.
// Block: 256 thr = 8 waves; MB=4 m-tiles per block (A slabs staged to LDS via
// CDNA5 async-to-LDS DMA), one n-tile per wave, 4 accumulators per wave so each
// B fetch feeds 4 v_wmma issues.
__global__ __launch_bounds__(256) void k_gemm(const _Float16* __restrict__ Ap,
                                              const _Float16* __restrict__ Bp,
                                              const float* __restrict__ bias,
                                              float* __restrict__ out,
                                              int ntiles, int Nreal) {
    __shared__ __align__(64) _Float16 Alds[MB * ABLK];   // 128 KB
    int mtile0 = blockIdx.y * MB;

    // Stage MB full-K A slabs into LDS with async DMA (no VGPR bounce).
    {
        const char* gsrc = (const char*)(Ap + (size_t)mtile0 * ABLK);
        unsigned ldsbase = (unsigned)(unsigned long long)(&Alds[0]);
        #pragma unroll
        for (int i = threadIdx.x; i < (MB * ABLK * 2) / 16; i += 256) {
            unsigned    loff = ldsbase + (unsigned)i * 16u;
            const char* g    = gsrc + (size_t)i * 16;
            asm volatile("global_load_async_to_lds_b128 %0, %1, off"
                         :: "v"(loff), "v"(g) : "memory");
        }
        asm volatile("s_wait_asynccnt 0x0" ::: "memory");
    }
    __syncthreads();

    int wave = threadIdx.x >> 5;
    int lane = threadIdx.x & 31;
    int ntile = blockIdx.x * 8 + wave;
    if (ntile >= ntiles) return;                   // wave-uniform; EXEC stays all-1s

    const _Float16* bbase = Bp + (size_t)ntile * ABLK + lane * 16;
    const _Float16* abase = Alds + lane * 16;

    v8f c[MB];
    #pragma unroll
    for (int mb = 0; mb < MB; ++mb) c[mb] = (v8f){};

    #pragma unroll 2
    for (int kt = 0; kt < KTILES; ++kt) {
        v16h b = *(const v16h*)(bbase + (size_t)kt * TILE_HALVES);
        __builtin_prefetch(bbase + (size_t)(kt + 4) * TILE_HALVES, 0, 3);
        #pragma unroll
        for (int mb = 0; mb < MB; ++mb) {
            v16h a = *(const v16h*)(abase + mb * ABLK + kt * TILE_HALVES);
            c[mb] = __builtin_amdgcn_wmma_f32_16x16x32_f16(false, a, false, b,
                                                           (short)0, c[mb], false, false);
        }
    }

    // C/D layout: lanes 0-15 -> M=p, lanes 16-31 -> M=8+p; N = lane%16
    int col = ntile * 16 + (lane & 15);
    if (col < Nreal) {
        float bv = bias[col];
        #pragma unroll
        for (int mb = 0; mb < MB; ++mb) {
            int mrow0 = (mtile0 + mb) * 16 + ((lane >> 4) << 3);
            #pragma unroll
            for (int p = 0; p < 8; ++p)
                out[(size_t)(mrow0 + p) * Nreal + col] = c[mb][p] + bv;
        }
    }
}

// ---------------- sequential LSTM scan ----------------
// One workgroup (32 waves), h/c in LDS, W_hh L2-resident. Writes h in packed-A f16.
__global__ __launch_bounds__(1024) void k_scan(const float* __restrict__ xg,
                                               const float* __restrict__ Wp,
                                               _Float16* __restrict__ Hp) {
    __shared__ float hl[EDIM];
    __shared__ float cl[EDIM];
    int r = threadIdx.x;
    hl[r] = 0.0f;
    cl[r] = 0.0f;
    __syncthreads();

    const float4* W = (const float4*)Wp;
    for (int t = 0; t < SEQ; ++t) {
        const float* x = xg + (size_t)t * GATE4;
        float a0 = x[r];
        float a1 = x[EDIM + r];
        float a2 = x[2 * EDIM + r];
        float a3 = x[3 * EDIM + r];
        #pragma unroll 4
        for (int k = 0; k < EDIM; ++k) {
            float hk = hl[k];                       // LDS broadcast
            float4 w = W[(size_t)k * EDIM + r];     // coalesced b128
            a0 = fmaf(w.x, hk, a0);
            a1 = fmaf(w.y, hk, a1);
            a2 = fmaf(w.z, hk, a2);
            a3 = fmaf(w.w, hk, a3);
        }
        float ig = 1.0f / (1.0f + __expf(-a0));
        float fg = 1.0f / (1.0f + __expf(-a1));
        float gg = tanhf(a2);
        float og = 1.0f / (1.0f + __expf(-a3));
        float cc = fg * cl[r] + ig * gg;
        float hh = og * tanhf(cc);
        __syncthreads();                            // all reads of h done
        cl[r] = cc;
        hl[r] = hh;
        Hp[a_pack_index(t >> 4, r >> 5, t & 15, r & 31)] = (_Float16)hh;
        __syncthreads();                            // writes visible for next step
    }
}

// ---------------- fused row-wise log_softmax (in place on d_out) ----------------
__global__ __launch_bounds__(256) void k_logsoftmax(float* __restrict__ out) {
    __shared__ float red[256];
    int row = blockIdx.x;
    float* p = out + (size_t)row * VOCAB;

    float m = -3.4e38f;
    for (int i = threadIdx.x; i < VOCAB; i += 256) m = fmaxf(m, p[i]);
    red[threadIdx.x] = m;
    __syncthreads();
    for (int s = 128; s > 0; s >>= 1) {
        if (threadIdx.x < s) red[threadIdx.x] = fmaxf(red[threadIdx.x], red[threadIdx.x + s]);
        __syncthreads();
    }
    m = red[0];
    __syncthreads();

    float sum = 0.0f;
    for (int i = threadIdx.x; i < VOCAB; i += 256) sum += __expf(p[i] - m);
    red[threadIdx.x] = sum;
    __syncthreads();
    for (int s = 128; s > 0; s >>= 1) {
        if (threadIdx.x < s) red[threadIdx.x] += red[threadIdx.x + s];
        __syncthreads();
    }
    float lse = m + __logf(red[0]);
    __syncthreads();

    for (int i = threadIdx.x; i < VOCAB; i += 256) p[i] -= lse;
}

// ---------------- launcher ----------------
extern "C" void kernel_launch(void* const* d_in, const int* in_sizes, int n_in,
                              void* d_out, int out_size, void* d_ws, size_t ws_size,
                              hipStream_t stream) {
    (void)in_sizes; (void)n_in; (void)out_size; (void)ws_size;
    const int*   sent   = (const int*)  d_in[0];
    const float* wte    = (const float*)d_in[1];
    const float* W_ih   = (const float*)d_in[2];
    const float* W_hh   = (const float*)d_in[3];
    const float* b_ih   = (const float*)d_in[4];
    const float* b_hh   = (const float*)d_in[5];
    const float* W_head = (const float*)d_in[6];
    const float* b_head = (const float*)d_in[7];
    float* out = (float*)d_out;

    char* ws = (char*)d_ws;
    size_t off = 0;
    auto alloc = [&](size_t bytes) -> void* {
        void* p = ws + off;
        off = (off + bytes + 255) & ~(size_t)255;
        return p;
    };
    float*    bsum = (float*)   alloc((size_t)GATE4 * 4);                 // 16 KB
    _Float16* ApX  = (_Float16*)alloc((size_t)MT_SEQ  * ABLK * 2);        // 4 MB
    _Float16* BpIH = (_Float16*)alloc((size_t)NT_WIH  * ABLK * 2);        // 8 MB
    float*    xg   = (float*)   alloc((size_t)SEQ * GATE4 * 4);           // 32 MB
    float*    Wp   = (float*)   alloc((size_t)EDIM * EDIM * 4 * 4);       // 16 MB
    _Float16* ApH  = (_Float16*)alloc((size_t)MT_SEQ  * ABLK * 2);        // 4 MB
    _Float16* BpHD = (_Float16*)alloc((size_t)NT_HEAD * ABLK * 2);        // ~103 MB

    k_bias_sum  <<<GATE4 / 256, 256, 0, stream>>>(b_ih, b_hh, bsum);
    k_pack_embed<<<(SEQ * EDIM) / 256, 256, 0, stream>>>(sent, wte, ApX);
    k_pack_B    <<<((size_t)NT_WIH  * ABLK) / 256, 256, 0, stream>>>(W_ih,   BpIH, GATE4);
    k_pack_B    <<<((size_t)NT_HEAD * ABLK) / 256, 256, 0, stream>>>(W_head, BpHD, VOCAB);
    k_pack_whh  <<<(EDIM * EDIM) / 256, 256, 0, stream>>>(W_hh, Wp);

    dim3 g1((NT_WIH + 7) / 8, MT_SEQ / MB);
    k_gemm<<<g1, 256, 0, stream>>>(ApX, BpIH, bsum, xg, NT_WIH, GATE4);

    k_scan<<<1, 1024, 0, stream>>>(xg, Wp, ApH);

    dim3 g2((NT_HEAD + 7) / 8, MT_SEQ / MB);
    k_gemm<<<g2, 256, 0, stream>>>(ApH, BpHD, b_head, out, NT_HEAD, VOCAB);

    k_logsoftmax<<<SEQ, 256, 0, stream>>>(out);
}
